// TensorProductAttention_60756607369546
// MI455X (gfx1250) — matrix operand order repaired and checked by reading, not compiled, not run on hardware
//
#include <hip/hip_runtime.h>
#include <hip/hip_bf16.h>

// ---------------------------------------------------------------------------
// Tensor Product Attention for MI455X (gfx1250, wave32, WMMA).
// B=2, S=2048, D=1024, H=16, d=64, Q_RANK=6, KV_RANK=2.
// ---------------------------------------------------------------------------

#define BATCH 2
#define SEQ 2048
#define DMODEL 1024
#define NHEADS 16
#define HDIM 64
#define QRANK 6
#define KVRANK 2
#define TOKENS (BATCH * SEQ)          // 4096
#define NPROJ 800                     // 96+384+32+128+32+128
#define NPAD 832                      // padded to multiple of 64

typedef __attribute__((ext_vector_type(16))) __bf16 bf16x16;
typedef __attribute__((ext_vector_type(8)))  __bf16 bf16x8;
typedef __attribute__((ext_vector_type(8)))  float  f32x8;

__device__ __forceinline__ __bf16 f2bf(float f) {
    union { float f; unsigned u; } v; v.f = f;
    unsigned r = v.u + 0x7FFFu + ((v.u >> 16) & 1u);   // round-to-nearest-even
    unsigned short h = (unsigned short)(r >> 16);
    return *reinterpret_cast<__bf16*>(&h);
}

__device__ __forceinline__ bf16x16 load_afrag(const __bf16* p) {
    // A-fragment 16x32: lane = row, elements 0..15 = K(+half*16) contiguous
    return *(const bf16x16*)(p);
}

// ---------------------------------------------------------------------------
// fp32 -> bf16 conversion / zero fill
// ---------------------------------------------------------------------------
__global__ void f32_to_bf16_kernel(const float* __restrict__ src,
                                   __bf16* __restrict__ dst, int n) {
    int i = blockIdx.x * blockDim.x + threadIdx.x;
    if (i < n) dst[i] = f2bf(src[i]);
}

__global__ void fill_zero_bf16_kernel(__bf16* __restrict__ dst, int n) {
    int i = blockIdx.x * blockDim.x + threadIdx.x;
    if (i < n) dst[i] = f2bf(0.0f);
}

// ---------------------------------------------------------------------------
// NT GEMM: C[m,n] = sum_k A[m,k] * B[n,k] (+ bias[n])
// A: MxK bf16 row-major, B: NxK bf16 row-major, C: MxN fp32.
// One wave produces a 32(M) x 64(N) tile: 2 A-frags x 4 B-frags = 8 WMMAs
// per 192B of per-lane fragment loads (A-frags used 4x, B-frags 2x).
// Prefetch is unconditional: gfx1250 speculative prefetches past the end
// of the stream are silently dropped, so no guard branch in the hot loop.
// ---------------------------------------------------------------------------
__global__ void gemm_bf16_nt(const __bf16* __restrict__ A,
                             const __bf16* __restrict__ B,
                             const float*  __restrict__ bias,
                             float* __restrict__ C,
                             int M, int N, int K) {
    const int wid  = (blockIdx.x * blockDim.x + threadIdx.x) >> 5;
    const int lane = threadIdx.x & 31;
    const int r    = lane & 15;
    const int half = lane >> 4;
    const int tilesN = N >> 6;
    const int wm = wid / tilesN;
    const int wn = wid % tilesN;
    if (wm * 32 >= M) return;                 // wave-uniform
    const int m0 = wm * 32, n0 = wn * 64;

    f32x8 acc[2][4];
    #pragma unroll
    for (int a = 0; a < 2; ++a)
        #pragma unroll
        for (int t = 0; t < 4; ++t) acc[a][t] = (f32x8){};

    const __bf16* ap0 = A + (size_t)(m0 + r) * K + half * 16;
    const __bf16* ap1 = ap0 + (size_t)16 * K;
    const __bf16* bp  = B + (size_t)(n0 + r) * K + half * 16;

    for (int k0 = 0; k0 < K; k0 += 32) {
        // keep the L2 -> WGP stream ~8 k-steps ahead of the WMMA pipe
        __builtin_prefetch(ap0 + k0 + 256, 0, 1);
        __builtin_prefetch(ap1 + k0 + 256, 0, 1);
        __builtin_prefetch(bp  + k0 + 256, 0, 1);

        bf16x16 af0 = load_afrag(ap0 + k0);
        bf16x16 af1 = load_afrag(ap1 + k0);
        // B fragments: 32x16, lane = N, 16 contiguous K per half
        bf16x16 bfr[4];
        #pragma unroll
        for (int t = 0; t < 4; ++t)
            bfr[t] = *(const bf16x16*)(bp + k0 + (size_t)(t * 16) * K);
        #pragma unroll
        for (int t = 0; t < 4; ++t) {
            acc[0][t] = __builtin_amdgcn_wmma_f32_16x16x32_bf16(false, af0, false, bfr[t], (short)0, acc[0][t], false, false);
            acc[1][t] = __builtin_amdgcn_wmma_f32_16x16x32_bf16(false, af1, false, bfr[t], (short)0, acc[1][t], false, false);
        }
    }

    // C layout: VGPR i -> row m0 + a*16 + half*8 + i ; col = n0 + t*16 + r
    #pragma unroll
    for (int a = 0; a < 2; ++a)
        #pragma unroll
        for (int t = 0; t < 4; ++t) {
            int col = n0 + t * 16 + r;
            float bs = bias ? bias[col] : 0.0f;
            #pragma unroll
            for (int i = 0; i < 8; ++i)
                C[(size_t)(m0 + a * 16 + half * 8 + i) * N + col] = acc[a][t][i] + bs;
        }
}

// ---------------------------------------------------------------------------
// build_qkv: per token, add biases, apply interleaved RoPE to b_q/b_k,
// contract over rank -> q,k,v. Writes Q,K as (B,H,S,64) bf16 and V
// transposed (B,H,64,S) bf16 so attention B-fragments are contiguous.
// P layout (stride NPAD): a_q[0,96) b_q[96,480) a_k[480,512) b_k[512,640)
//                         a_v[640,672) b_v[672,800)
// ---------------------------------------------------------------------------
__global__ void build_qkv(const float* __restrict__ P,
                          const float* __restrict__ baq, const float* __restrict__ bbq,
                          const float* __restrict__ bak, const float* __restrict__ bbk,
                          const float* __restrict__ bav, const float* __restrict__ bbv,
                          const float* __restrict__ cosb, const float* __restrict__ sinb,
                          __bf16* __restrict__ Qo, __bf16* __restrict__ Ko,
                          __bf16* __restrict__ VTo) {
    __shared__ float aq[NHEADS * QRANK];   // 96
    __shared__ float ak[NHEADS * KVRANK];  // 32
    __shared__ float av[NHEADS * KVRANK];  // 32

    const int tok = blockIdx.x;            // 0..4095
    const int b = tok >> 11;
    const int s = tok & (SEQ - 1);
    const int d = threadIdx.x;             // 0..63
    const float* row = P + (size_t)tok * NPAD;

    for (int i = d; i < NHEADS * QRANK; i += 64) aq[i] = row[i] + baq[i];
    if (d < 32) {
        ak[d] = row[480 + d] + bak[d];
        av[d] = row[640 + d] + bav[d];
    }
    __syncthreads();

    const int  j   = d >> 1;
    const float c  = cosb[s * (HDIM / 2) + j];
    const float sn = sinb[s * (HDIM / 2) + j];
    const bool odd = d & 1;
    const int  e   = d & ~1;

    float bq[QRANK], bk[KVRANK], bv[KVRANK];
    #pragma unroll
    for (int rr = 0; rr < QRANK; ++rr) {
        float x1 = row[96 + rr * HDIM + e]     + bbq[rr * HDIM + e];
        float x2 = row[96 + rr * HDIM + e + 1] + bbq[rr * HDIM + e + 1];
        bq[rr] = odd ? (x1 * sn + x2 * c) : (x1 * c - x2 * sn);
    }
    #pragma unroll
    for (int rr = 0; rr < KVRANK; ++rr) {
        float x1 = row[512 + rr * HDIM + e]     + bbk[rr * HDIM + e];
        float x2 = row[512 + rr * HDIM + e + 1] + bbk[rr * HDIM + e + 1];
        bk[rr] = odd ? (x1 * sn + x2 * c) : (x1 * c - x2 * sn);
        bv[rr] = row[672 + rr * HDIM + d] + bbv[rr * HDIM + d];
    }

    #pragma unroll
    for (int h = 0; h < NHEADS; ++h) {
        float qv = 0.0f;
        #pragma unroll
        for (int rr = 0; rr < QRANK; ++rr) qv += aq[h * QRANK + rr] * bq[rr];
        float kv = ak[h * 2] * bk[0] + ak[h * 2 + 1] * bk[1];
        float vv = av[h * 2] * bv[0] + av[h * 2 + 1] * bv[1];
        size_t bh = (size_t)(b * NHEADS + h);
        Qo[(bh * SEQ + s) * HDIM + d]  = f2bf(qv);
        Ko[(bh * SEQ + s) * HDIM + d]  = f2bf(kv);
        VTo[(bh * HDIM + d) * SEQ + s] = f2bf(vv);
    }
}

// ---------------------------------------------------------------------------
// Causal flash attention, one wave per (b,h, 16-query tile).
// Online softmax; P tile goes through LDS to become an A-fragment (LDS ops
// from one wave are in-order on CDNA5, no barrier needed).
// Writes attn (B*S, 1024) bf16, col = h*64 + d.
// ---------------------------------------------------------------------------
__global__ void flash_attn(const __bf16* __restrict__ Q,
                           const __bf16* __restrict__ Kb,
                           const __bf16* __restrict__ Vt,
                           __bf16* __restrict__ attn) {
    __shared__ __bf16 plds[4][16 * 32];

    const int widL = threadIdx.x >> 5;               // wave in block
    const int wid  = blockIdx.x * 4 + widL;
    const int lane = threadIdx.x & 31;
    const int r    = lane & 15;
    const int half = lane >> 4;

    const int qt = wid & 127;                        // S/16 = 128 q-tiles
    const int bh = wid >> 7;                         // 0..31
    const int qbase = qt * 16;

    const __bf16* qp = Q  + (size_t)bh * SEQ * HDIM;
    const __bf16* kp = Kb + (size_t)bh * SEQ * HDIM;
    const __bf16* vp = Vt + (size_t)bh * HDIM * SEQ;

    // Q fragments (d 0..31 and 32..63), loaded once
    const __bf16* qrow = qp + (size_t)(qbase + r) * HDIM + half * 16;
    bf16x16 qf0 = load_afrag(qrow);
    bf16x16 qf1 = load_afrag(qrow + 32);

    float mrow[8], lrow[8];
    f32x8 acc[4] = {};
    #pragma unroll
    for (int i = 0; i < 8; ++i) { mrow[i] = -1e30f; lrow[i] = 0.0f; }
    const float scale = 0.125f;                      // 1/sqrt(64)

    const int nk = (qbase + 16 + 31) >> 5;           // 32-key steps (causal)
    for (int t = 0; t < nk; ++t) {
        const int kk0 = t * 32;
        float s0[8], s1[8];
        #pragma unroll
        for (int u = 0; u < 2; ++u) {
            const int n0 = kk0 + u * 16;
            const __bf16* krow = kp + (size_t)(n0 + r) * HDIM + half * 16;
            bf16x16 kf0 = *(const bf16x16*)(krow);
            bf16x16 kf1 = *(const bf16x16*)(krow + 32);
            f32x8 sc = {};
            sc = __builtin_amdgcn_wmma_f32_16x16x32_bf16(false, qf0, false, kf0, (short)0, sc, false, false);
            sc = __builtin_amdgcn_wmma_f32_16x16x32_bf16(false, qf1, false, kf1, (short)0, sc, false, false);
            const int key = n0 + r;
            float* s = u ? s1 : s0;
            #pragma unroll
            for (int i = 0; i < 8; ++i) {
                int qidx = qbase + half * 8 + i;
                float v = sc[i] * scale;
                s[i] = (key <= qidx) ? v : -1e30f;
            }
        }
        // per-row (M) reductions across the 16 lanes that hold the row
        float tm[8];
        #pragma unroll
        for (int i = 0; i < 8; ++i) tm[i] = fmaxf(s0[i], s1[i]);
        #pragma unroll
        for (int m = 1; m < 16; m <<= 1)
            #pragma unroll
            for (int i = 0; i < 8; ++i) tm[i] = fmaxf(tm[i], __shfl_xor(tm[i], m, 32));

        float corr[8], ts[8];
        #pragma unroll
        for (int i = 0; i < 8; ++i) {
            float nm = fmaxf(mrow[i], tm[i]);
            corr[i]  = __expf(mrow[i] - nm);
            mrow[i]  = nm;
            s0[i] = __expf(s0[i] - nm);
            s1[i] = __expf(s1[i] - nm);
            ts[i] = s0[i] + s1[i];
        }
        #pragma unroll
        for (int m = 1; m < 16; m <<= 1)
            #pragma unroll
            for (int i = 0; i < 8; ++i) ts[i] += __shfl_xor(ts[i], m, 32);
        #pragma unroll
        for (int i = 0; i < 8; ++i) lrow[i] = lrow[i] * corr[i] + ts[i];
        #pragma unroll
        for (int dt = 0; dt < 4; ++dt)
            #pragma unroll
            for (int i = 0; i < 8; ++i) acc[dt][i] *= corr[i];

        // stage P (16x32) to LDS in row-major, reload as A-fragment
        __bf16* pl = plds[widL];
        #pragma unroll
        for (int i = 0; i < 8; ++i) {
            pl[(half * 8 + i) * 32 + r]      = f2bf(s0[i]);
            pl[(half * 8 + i) * 32 + 16 + r] = f2bf(s1[i]);
        }
        bf16x16 pf = load_afrag(pl + r * 32 + half * 16);
        #pragma unroll
        for (int dt = 0; dt < 4; ++dt) {
            const __bf16* vrow = vp + (size_t)(dt * 16 + r) * SEQ + kk0 + half * 16;
            bf16x16 vf = *(const bf16x16*)(vrow);
            acc[dt] = __builtin_amdgcn_wmma_f32_16x16x32_bf16(false, pf, false, vf, (short)0, acc[dt], false, false);
        }
    }

    // finalize and store
    const int b = bh >> 4, h = bh & 15;
    float inv[8];
    #pragma unroll
    for (int i = 0; i < 8; ++i) inv[i] = 1.0f / lrow[i];
    #pragma unroll
    for (int dt = 0; dt < 4; ++dt)
        #pragma unroll
        for (int i = 0; i < 8; ++i) {
            size_t rowi = (size_t)(b * SEQ + qbase + half * 8 + i);
            attn[rowi * DMODEL + h * HDIM + dt * 16 + r] = f2bf(acc[dt][i] * inv[i]);
        }
}

// ---------------------------------------------------------------------------
// Launch
// ---------------------------------------------------------------------------
extern "C" void kernel_launch(void* const* d_in, const int* in_sizes, int n_in,
                              void* d_out, int out_size, void* d_ws, size_t ws_size,
                              hipStream_t stream) {
    const float* x    = (const float*)d_in[0];
    const float* Waq  = (const float*)d_in[1];
    const float* baq  = (const float*)d_in[2];
    const float* Wbq  = (const float*)d_in[3];
    const float* bbq  = (const float*)d_in[4];
    const float* Wak  = (const float*)d_in[5];
    const float* bak  = (const float*)d_in[6];
    const float* Wbk  = (const float*)d_in[7];
    const float* bbk  = (const float*)d_in[8];
    const float* Wav  = (const float*)d_in[9];
    const float* bav  = (const float*)d_in[10];
    const float* Wbv  = (const float*)d_in[11];
    const float* bbv  = (const float*)d_in[12];
    const float* Wo   = (const float*)d_in[13];
    const float* bo   = (const float*)d_in[14];
    const float* cosb = (const float*)d_in[15];
    const float* sinb = (const float*)d_in[16];

    // workspace carve-out (256B aligned)
    char* w = (char*)d_ws;
    size_t off = 0;
    auto carve = [&](size_t bytes) {
        void* p = w + off;
        off += (bytes + 255) & ~(size_t)255;
        return p;
    };
    __bf16* xbf   = (__bf16*)carve((size_t)TOKENS * DMODEL * 2);   // 8 MB
    __bf16* wcat  = (__bf16*)carve((size_t)NPAD * DMODEL * 2);     // 1.7 MB
    __bf16* wobf  = (__bf16*)carve((size_t)DMODEL * DMODEL * 2);   // 2 MB
    float*  Pbuf  = (float*) carve((size_t)TOKENS * NPAD * 4);     // 13.6 MB
    __bf16* Qbf   = (__bf16*)carve((size_t)BATCH * NHEADS * SEQ * HDIM * 2);
    __bf16* Kbf   = (__bf16*)carve((size_t)BATCH * NHEADS * SEQ * HDIM * 2);
    __bf16* VTbf  = (__bf16*)carve((size_t)BATCH * NHEADS * HDIM * SEQ * 2);
    __bf16* Abf   = (__bf16*)carve((size_t)TOKENS * DMODEL * 2);   // attn

    const int CT = 256;
    auto cdiv = [](int a, int b) { return (a + b - 1) / b; };

    // 1) convert x + weights to bf16 (weights concatenated + zero-padded)
    f32_to_bf16_kernel<<<cdiv(TOKENS * DMODEL, CT), CT, 0, stream>>>(x, xbf, TOKENS * DMODEL);
    f32_to_bf16_kernel<<<cdiv(96 * 1024, CT),  CT, 0, stream>>>(Waq, wcat + (size_t)0   * 1024, 96 * 1024);
    f32_to_bf16_kernel<<<cdiv(384 * 1024, CT), CT, 0, stream>>>(Wbq, wcat + (size_t)96  * 1024, 384 * 1024);
    f32_to_bf16_kernel<<<cdiv(32 * 1024, CT),  CT, 0, stream>>>(Wak, wcat + (size_t)480 * 1024, 32 * 1024);
    f32_to_bf16_kernel<<<cdiv(128 * 1024, CT), CT, 0, stream>>>(Wbk, wcat + (size_t)512 * 1024, 128 * 1024);
    f32_to_bf16_kernel<<<cdiv(32 * 1024, CT),  CT, 0, stream>>>(Wav, wcat + (size_t)640 * 1024, 32 * 1024);
    f32_to_bf16_kernel<<<cdiv(128 * 1024, CT), CT, 0, stream>>>(Wbv, wcat + (size_t)672 * 1024, 128 * 1024);
    fill_zero_bf16_kernel<<<cdiv(32 * 1024, CT), CT, 0, stream>>>(wcat + (size_t)800 * 1024, 32 * 1024);
    f32_to_bf16_kernel<<<cdiv(DMODEL * DMODEL, CT), CT, 0, stream>>>(Wo, wobf, DMODEL * DMODEL);

    // 2) fused projection GEMM: P = Xbf @ Wcat^T   (4096 x 832, K=1024)
    {
        int waves = (TOKENS / 32) * (NPAD / 64);     // 1664
        gemm_bf16_nt<<<waves / 8, 256, 0, stream>>>(xbf, wcat, nullptr, Pbuf,
                                                    TOKENS, NPAD, DMODEL);
    }

    // 3) RoPE + rank contraction -> Q, K, V^T (bf16)
    build_qkv<<<TOKENS, 64, 0, stream>>>(Pbuf, baq, bbq, bak, bbk, bav, bbv,
                                         cosb, sinb, Qbf, Kbf, VTbf);

    // 4) causal flash attention (WMMA) -> attn bf16 (4096 x 1024)
    {
        int waves = BATCH * NHEADS * (SEQ / 16);     // 4096
        flash_attn<<<waves / 4, 128, 0, stream>>>(Qbf, Kbf, VTbf, Abf);
    }

    // 5) output projection: out = attn @ Wo^T + bo  (fp32)
    {
        int waves = (TOKENS / 32) * (DMODEL / 64);   // 2048
        gemm_bf16_nt<<<waves / 8, 256, 0, stream>>>(Abf, wobf, bo, (float*)d_out,
                                                    TOKENS, DMODEL, DMODEL);
    }
}
